// BaselineOut_60705067761914
// MI455X (gfx1250) — matrix operand order repaired and checked by reading, not compiled, not run on hardware
//
#include <hip/hip_runtime.h>
#include <hip/hip_bf16.h>

typedef unsigned int   u32;
typedef unsigned short u16;

typedef __attribute__((ext_vector_type(16))) __bf16 v16bf;
typedef __attribute__((ext_vector_type(8)))  float  v8f;

union Frag16 { v16bf v; uint4 q[2]; };

#define BB   32
#define LA   2048
#define LQ   64
#define LO   32
#define HH   1024
#define NOPT 5

__device__ __forceinline__ u32 f2bf(float x) {
    u32 u = __float_as_uint(x);
    return (u + 0x7FFFu + ((u >> 16) & 1u)) >> 16;   // round-to-nearest-even
}

#if __has_builtin(__builtin_amdgcn_tanhf)
__device__ __forceinline__ float fast_tanh(float x) { return __builtin_amdgcn_tanhf(x); }
#else
__device__ __forceinline__ float fast_tanh(float x) { return tanhf(x); }
#endif

// ---------------------------------------------------------------------------
// fp32 -> bf16 weight conversion
// ---------------------------------------------------------------------------
__global__ __launch_bounds__(256) void cvt_bf16_kernel(const float* __restrict__ in,
                                                       u16* __restrict__ out, int n) {
    int i = blockIdx.x * 256 + threadIdx.x;
    if (i < n) out[i] = (u16)f2bf(in[i]);
}

// ---------------------------------------------------------------------------
// gather oqc[b] = question[b][answer_indices[b]]
// ---------------------------------------------------------------------------
__global__ __launch_bounds__(256) void gather_kernel(const float* __restrict__ q,
                                                     const int* __restrict__ idx,
                                                     float* __restrict__ oqc) {
    int b = blockIdx.x;
    const float* src = q + ((long)b * LQ + idx[b]) * HH;
    for (int i = threadIdx.x; i < HH; i += 256) oqc[(long)b * HH + i] = src[i];
}

// ---------------------------------------------------------------------------
// small fp32 projection: Y[r][n] = sum_k X[r][k]*W[n][k] + bias[n]
// ---------------------------------------------------------------------------
__global__ __launch_bounds__(256) void proj_kernel(const float* __restrict__ X,
                                                   const float* __restrict__ W,
                                                   const float* __restrict__ bias,
                                                   float* __restrict__ Y, int total) {
    int gid = blockIdx.x * 256 + threadIdx.x;
    if (gid >= total) return;
    int r = gid >> 10, n = gid & 1023;
    const float* x  = X + (long)r * HH;
    const float* wr = W + (long)n * HH;
    float acc = 0.f;
    for (int k = 0; k < HH; k += 4) {
        float4 a = *(const float4*)(x + k);
        float4 b = *(const float4*)(wr + k);
        acc += a.x * b.x + a.y * b.y + a.z * b.z + a.w * b.w;
    }
    Y[gid] = acc + bias[n];
}

// ---------------------------------------------------------------------------
// WMMA fused score kernel (TSM rows per WG):
//   s[row] = vb + sum_n vw[n] * tanh( A[row,:]·W[n,:] + Qp[qrow][n] + Kb[n] )
// 8 waves each own a 16-wide N slice of each 128-wide N chunk; K steps of 32.
// ---------------------------------------------------------------------------
template <int TSM>
__global__ __launch_bounds__(256) void score_wmma_kernel(
        const float* __restrict__ A,      // [Mtotal][1024] fp32
        const u16*   __restrict__ Wb,     // [1024][1024] bf16
        const float* __restrict__ Qp,     // [nq][1024]
        const float* __restrict__ Kbias,  // [1024]
        const float* __restrict__ vw,     // [1024]
        const float* __restrict__ vbias,  // [1]
        float* __restrict__ scores,       // [Mtotal]
        int rows_per_q)
{
    constexpr int MS = TSM / 16;                    // M-subtiles per WG
    __shared__ __align__(16) u16 Asm[TSM * 1032];   // padded stride: conflict-free
    __shared__ float red[TSM];

    const int tid  = threadIdx.x;
    const int wave = tid >> 5;
    const int lane = tid & 31;
    const int row0 = blockIdx.x * TSM;

    // stage A tile (TSM x 1024 fp32 -> bf16 in LDS)
    for (int i = tid; i < TSM * 256; i += 256) {
        int r = i >> 8;
        int k = (i & 255) << 2;
        float4 f = *(const float4*)(A + (long)(row0 + r) * HH + k);
        uint2 p;
        p.x = f2bf(f.x) | (f2bf(f.y) << 16);
        p.y = f2bf(f.z) | (f2bf(f.w) << 16);
        *(uint2*)&Asm[r * 1032 + k] = p;
    }
    if (tid < TSM) red[tid] = 0.0f;
    __syncthreads();

    const int qrow = row0 / rows_per_q;
    const float* qp = Qp + (long)qrow * HH;

    const int hi    = lane >> 4;    // half-wave select
    const int lo16  = lane & 15;
    const int koffA = hi * 16;      // A-frag: halves split K 0..15/16..31 (matches B)

    float acc[MS * 8];
#pragma unroll
    for (int j = 0; j < MS * 8; ++j) acc[j] = 0.f;

    const v8f vzero = {0.f, 0.f, 0.f, 0.f, 0.f, 0.f, 0.f, 0.f};

#pragma unroll 1
    for (int nc = 0; nc < 8; ++nc) {
        const int n = nc * 128 + wave * 16 + lo16;
        const u16* wrow = Wb + (long)n * HH;
        v8f c[MS];
#pragma unroll
        for (int s = 0; s < MS; ++s) c[s] = vzero;

#pragma clang loop unroll_count(2)
        for (int kb = 0; kb < HH; kb += 32) {
            Frag16 bfrag;
            // B: lane holds 16 contiguous K of column n (halves split K 0..15/16..31)
            const u16* wp = wrow + kb + hi * 16;
            bfrag.q[0] = *(const uint4*)(wp);
            bfrag.q[1] = *(const uint4*)(wp + 8);
#pragma unroll
            for (int s = 0; s < MS; ++s) {
                Frag16 a;
                const u16* ap = &Asm[(lo16 + 16 * s) * 1032 + kb + koffA];
                a.q[0] = *(const uint4*)(ap);
                a.q[1] = *(const uint4*)(ap + 8);
                c[s] = __builtin_amdgcn_wmma_f32_16x16x32_bf16(false, a.v, false, bfrag.v,
                                                               (short)0, c[s], false, false);
            }
        }
        const float qv = qp[n] + Kbias[n];
        const float vn = vw[n];
#pragma unroll
        for (int s = 0; s < MS; ++s) {
#pragma unroll
            for (int j = 0; j < 8; ++j) acc[s * 8 + j] += vn * fast_tanh(c[s][j] + qv);
        }
    }

    // butterfly sum across the 16 lanes of each half-wave (same row set, distinct n)
#pragma unroll
    for (int m = 8; m >= 1; m >>= 1) {
#pragma unroll
        for (int j = 0; j < MS * 8; ++j) acc[j] += __shfl_xor(acc[j], m, 32);
    }
    if (lo16 == 0) {
#pragma unroll
        for (int s = 0; s < MS; ++s) {
#pragma unroll
            for (int j = 0; j < 8; ++j)
                atomicAdd(&red[s * 16 + j + 8 * hi], acc[s * 8 + j]);
        }
    }
    __syncthreads();
    if (tid < TSM) scores[row0 + tid] = red[tid] + vbias[0];
}

// ---------------------------------------------------------------------------
// softmax over groups of length L (one block per group)
// ---------------------------------------------------------------------------
__global__ __launch_bounds__(256) void softmax_kernel(const float* __restrict__ s,
                                                      float* __restrict__ w, int L) {
    int g = blockIdx.x;
    const float* x = s + (long)g * L;
    float* y = w + (long)g * L;
    __shared__ float sm[256];
    int t = threadIdx.x;
    float m = -1e30f;
    for (int i = t; i < L; i += 256) m = fmaxf(m, x[i]);
    sm[t] = m; __syncthreads();
    for (int o = 128; o > 0; o >>= 1) { if (t < o) sm[t] = fmaxf(sm[t], sm[t + o]); __syncthreads(); }
    m = sm[0]; __syncthreads();
    float sum = 0.f;
    for (int i = t; i < L; i += 256) sum += __expf(x[i] - m);
    sm[t] = sum; __syncthreads();
    for (int o = 128; o > 0; o >>= 1) { if (t < o) sm[t] += sm[t + o]; __syncthreads(); }
    float inv = 1.0f / sm[0];
    for (int i = t; i < L; i += 256) y[i] = __expf(x[i] - m) * inv;
}

__global__ __launch_bounds__(256) void zero_kernel(float* __restrict__ p, int n) {
    int i = blockIdx.x * 256 + threadIdx.x;
    if (i < n) p[i] = 0.f;
}

// stage-1 weighted sum: out[b][h] += sum_l w[b][l] * V[b][l][h]  (l split across grid.y)
__global__ __launch_bounds__(256) void wsum_atomic_kernel(const float* __restrict__ V,
                                                          const float* __restrict__ w,
                                                          float* __restrict__ out) {
    int b = blockIdx.x, part = blockIdx.y;
    const int Lc = LA / 8;
    const float* vb = V + ((long)b * LA + (long)part * Lc) * HH;
    const float* wb = w + (long)b * LA + part * Lc;
    int t = threadIdx.x;
    float4 acc = {0.f, 0.f, 0.f, 0.f};
    for (int l = 0; l < Lc; ++l) {
        float wl = wb[l];
        float4 v = *(const float4*)(vb + (long)l * HH + t * 4);
        acc.x += wl * v.x; acc.y += wl * v.y; acc.z += wl * v.z; acc.w += wl * v.w;
    }
    float* o = out + (long)b * HH + t * 4;
    atomicAdd(o + 0, acc.x); atomicAdd(o + 1, acc.y);
    atomicAdd(o + 2, acc.z); atomicAdd(o + 3, acc.w);
}

// stage-2 weighted sum: out[g][h] = sum_{l<32} w[g][l] * V[g][l][h]
__global__ __launch_bounds__(256) void wsum_direct_kernel(const float* __restrict__ V,
                                                          const float* __restrict__ w,
                                                          float* __restrict__ out) {
    int g = blockIdx.x;
    const float* vg = V + (long)g * LO * HH;
    const float* wg = w + (long)g * LO;
    int t = threadIdx.x;
    float4 acc = {0.f, 0.f, 0.f, 0.f};
    for (int l = 0; l < LO; ++l) {
        float wl = wg[l];
        float4 v = *(const float4*)(vg + (long)l * HH + t * 4);
        acc.x += wl * v.x; acc.y += wl * v.y; acc.z += wl * v.z; acc.w += wl * v.w;
    }
    *(float4*)(out + (long)g * HH + t * 4) = acc;
}

// logits[b][c] = feats[b][0:5120] . f_w[c] + f_b[c]
__global__ __launch_bounds__(256) void final_kernel(const float* __restrict__ feats,
                                                    const float* __restrict__ fw,
                                                    const float* __restrict__ fb,
                                                    float* __restrict__ out) {
    int t = blockIdx.x * 256 + threadIdx.x;
    if (t >= BB * NOPT) return;
    int b = t / NOPT, c = t % NOPT;
    const float* x  = feats + (long)b * NOPT * HH;
    const float* wr = fw + (long)c * NOPT * HH;
    float acc = 0.f;
    for (int k = 0; k < NOPT * HH; k += 4) {
        float4 a = *(const float4*)(x + k);
        float4 w = *(const float4*)(wr + k);
        acc += a.x * w.x + a.y * w.y + a.z * w.z + a.w * w.w;
    }
    out[t] = acc + fb[c];
}

// ---------------------------------------------------------------------------
extern "C" void kernel_launch(void* const* d_in, const int* in_sizes, int n_in,
                              void* d_out, int out_size, void* d_ws, size_t ws_size,
                              hipStream_t stream) {
    const float* article  = (const float*)d_in[0];
    const float* question = (const float*)d_in[1];
    const float* options  = (const float*)d_in[2];
    const int*   ans      = (const int*)d_in[3];
    const float* a_Qw = (const float*)d_in[4];
    const float* a_Qb = (const float*)d_in[5];
    const float* a_Kw = (const float*)d_in[6];
    const float* a_Kb = (const float*)d_in[7];
    const float* a_Vw = (const float*)d_in[8];
    const float* a_Vb = (const float*)d_in[9];
    const float* a_vw = (const float*)d_in[10];
    const float* a_vb = (const float*)d_in[11];
    const float* d_Qw = (const float*)d_in[12];
    const float* d_Qb = (const float*)d_in[13];
    const float* d_Kw = (const float*)d_in[14];
    const float* d_Kb = (const float*)d_in[15];
    const float* d_Vw = (const float*)d_in[16];
    const float* d_Vb = (const float*)d_in[17];
    const float* d_vw = (const float*)d_in[18];
    const float* d_vb = (const float*)d_in[19];
    const float* f_w  = (const float*)d_in[20];
    const float* f_b  = (const float*)d_in[21];
    float* out = (float*)d_out;

    char* ws = (char*)d_ws;
    const size_t MB = 1024 * 1024;
    u16*   Kw1b   = (u16*)(ws);                           // 2 MB
    u16*   Kw2b   = (u16*)(ws + 2 * MB);                  // 2 MB
    float* oqc    = (float*)(ws + 4 * MB);                // 128 KB
    float* Qp1    = (float*)(ws + 4 * MB + 128 * 1024);   // 128 KB
    float* sc1    = (float*)(ws + 4 * MB + 256 * 1024);   // 256 KB
    float* w1     = (float*)(ws + 4 * MB + 512 * 1024);   // 256 KB
    float* aqraw  = (float*)(ws + 4 * MB + 768 * 1024);   // 128 KB
    float* aq     = (float*)(ws + 4 * MB + 896 * 1024);   // 128 KB
    float* Qp2    = (float*)(ws + 5 * MB);                // 128 KB
    float* sc2    = (float*)(ws + 5 * MB + 128 * 1024);   // 32 KB
    float* w2     = (float*)(ws + 5 * MB + 160 * 1024);   // 32 KB
    float* foraw  = (float*)(ws + 5 * MB + 192 * 1024);   // 640 KB
    float* feats  = (float*)(ws + 5 * MB + 832 * 1024);   // 640 KB

    // weight conversion
    cvt_bf16_kernel<<<(HH * HH + 255) / 256, 256, 0, stream>>>(a_Kw, Kw1b, HH * HH);
    cvt_bf16_kernel<<<(HH * HH + 255) / 256, 256, 0, stream>>>(d_Kw, Kw2b, HH * HH);

    // stage 1: Qp1 = oqc @ a_Qw^T + a_Qb
    gather_kernel<<<BB, 256, 0, stream>>>(question, ans, oqc);
    proj_kernel<<<(BB * HH + 255) / 256, 256, 0, stream>>>(oqc, a_Qw, a_Qb, Qp1, BB * HH);

    // stage 1 fused scores over article (the big WMMA GEMM), 64-row tiles
    score_wmma_kernel<64><<<(BB * LA) / 64, 256, 0, stream>>>(article, Kw1b, Qp1, a_Kb,
                                                              a_vw, a_vb, sc1, LA);
    softmax_kernel<<<BB, 256, 0, stream>>>(sc1, w1, LA);

    zero_kernel<<<(BB * HH + 255) / 256, 256, 0, stream>>>(aqraw, BB * HH);
    wsum_atomic_kernel<<<dim3(BB, 8), 256, 0, stream>>>(article, w1, aqraw);

    // aq = aqraw @ a_Vw^T + a_Vb ; Qp2 = aq @ d_Qw^T + d_Qb
    proj_kernel<<<(BB * HH + 255) / 256, 256, 0, stream>>>(aqraw, a_Vw, a_Vb, aq, BB * HH);
    proj_kernel<<<(BB * HH + 255) / 256, 256, 0, stream>>>(aq, d_Qw, d_Qb, Qp2, BB * HH);

    // stage 2 fused scores over options, 32-row tiles (160-row q-groups)
    score_wmma_kernel<32><<<(BB * NOPT * LO) / 32, 256, 0, stream>>>(options, Kw2b, Qp2, d_Kb,
                                                                     d_vw, d_vb, sc2, NOPT * LO);
    softmax_kernel<<<BB * NOPT, 256, 0, stream>>>(sc2, w2, LO);
    wsum_direct_kernel<<<BB * NOPT, 256, 0, stream>>>(options, w2, foraw);

    // feats = foraw @ d_Vw^T + d_Vb ; logits = feats @ f_w^T + f_b
    proj_kernel<<<(BB * NOPT * HH + 255) / 256, 256, 0, stream>>>(foraw, d_Vw, d_Vb, feats,
                                                                  BB * NOPT * HH);
    final_kernel<<<1, 256, 0, stream>>>(feats, f_w, f_b, out);
}
